// cross_involution_36232344109350
// MI455X (gfx1250) — compile-verified, matrix operand order checked by reading, not run on hardware
//
#include <hip/hip_runtime.h>
#include <hip/hip_bf16.h>

// ---- problem constants -----------------------------------------------------
#define BATCH   4
#define C_IN    256
#define CR      64          // reduced channels after conv1
#define H_DIM   64
#define W_DIM   64
#define KSZ     7
#define KK      49          // 7*7
#define GROUPS  16
#define GC      16          // channels per group
#define K2G     784         // 49*16 conv2 out channels
#define TILE    16          // pixels per block (one row segment)
#define BN_EPS  1e-5f

typedef __bf16        v16bf __attribute__((ext_vector_type(16)));
typedef float         v8f   __attribute__((ext_vector_type(8)));
typedef unsigned int  v4u   __attribute__((ext_vector_type(4)));

// One WMMA operand fragment: 16 bf16 per lane = 32 contiguous bytes.
union Frag16 {
    unsigned short u[16];
    v4u            q[2];
    v16bf          v;
};

// pack two floats into one dword of bf16 via native fptrunc
union PackBF {
    __bf16       h[2];
    unsigned int u;
};
__device__ __forceinline__ unsigned int pack_bf2(float a, float b) {
    PackBF p;
    p.h[0] = (__bf16)a;
    p.h[1] = (__bf16)b;
    return p.u;
}
__device__ __forceinline__ unsigned short f2bf(float f) {
    PackBF p;
    p.h[0] = (__bf16)f;
    p.h[1] = (__bf16)0.0f;
    return (unsigned short)(p.u & 0xFFFFu);
}

// ISA 16-bit operand swizzle: k-within-chunk -> (lane-half, element index)
__device__ __forceinline__ int swz_hk(int kk) { return (kk >> 3) & 1; }
__device__ __forceinline__ int swz_i(int kk)  { return ((kk >> 4) << 3) + (kk & 7); }

// ---- prep: fold BN into w1, convert to bf16, fragment-major layout --------
// w1f: [mt(4)][kc(8)][lane(32)][i(16)]  ushort   (rows pre-scaled by BN inv)
// w2f: [mt(49)][kc(2)][lane(32)][i(16)] ushort
// bias1: [64] float  = beta - mean*inv
__global__ __launch_bounds__(256) void prep_weights_swz(
    const float* __restrict__ w1, const float* __restrict__ w2,
    const float* __restrict__ bn_gamma, const float* __restrict__ bn_beta,
    const float* __restrict__ bn_mean,  const float* __restrict__ bn_var,
    unsigned short* __restrict__ w1f, unsigned short* __restrict__ w2f,
    float* __restrict__ bias1) {
    int n = blockIdx.x * blockDim.x + threadIdx.x;
    if (n < CR * C_IN) {
        int m = n >> 8, k = n & 255;
        float inv = bn_gamma[m] * rsqrtf(bn_var[m] + BN_EPS);
        int mt = m >> 4, row = m & 15;
        int kc = k >> 5, kk = k & 31;
        w1f[(((mt * 8 + kc) * 32) + swz_hk(kk) * 16 + row) * 16 + swz_i(kk)] =
            f2bf(w1[n] * inv);
        if (k == 0) bias1[m] = bn_beta[m] - bn_mean[m] * inv;
    }
    if (n < K2G * CR) {
        int m = n >> 6, k = n & 63;
        int mt = m >> 4, row = m & 15;
        int kc = k >> 5, kk = k & 31;
        w2f[(((mt * 2 + kc) * 32) + swz_hk(kk) * 16 + row) * 16 + swz_i(kk)] =
            f2bf(w2[n]);
    }
}

// ---- fused conv1(+BN,ReLU) + conv2 + dynamic involution + residual --------
__global__ __launch_bounds__(256) void cross_involution_fused(
    const float* __restrict__ fm,            // [B,256,H,W]
    const float* __restrict__ guide,         // [B,256,H,W]
    const unsigned short* __restrict__ w1f,  // swizzled bf16 (BN-folded)
    const float* __restrict__ bias1,         // [64]
    const unsigned short* __restrict__ w2f,  // swizzled bf16
    const float* __restrict__ b2,            // [784]
    float* __restrict__ out) {               // [B,256,H,W]

    // fragment-major LDS: [kc][lane][16] ushort
    __shared__ __align__(16) unsigned short sGf[8 * 32 * 16];  // 8 KB
    __shared__ __align__(16) unsigned short sXf[2 * 32 * 16];  // 2 KB
    __shared__ float sW[K2G * TILE];                           // 49 KB

    const int tiles_x = W_DIM / TILE;            // 4
    int blk = blockIdx.x;
    int x0  = (blk % tiles_x) * TILE;
    int y   = (blk / tiles_x) % H_DIM;
    int b   = blk / (tiles_x * H_DIM);

    int t    = threadIdx.x;
    int wave = t >> 5;
    int lane = t & 31;
    int ln   = lane & 15;        // M-row (A) / N-col (B,C)
    int hk   = lane >> 4;        // K-half selector

    // ---- stage guide tile directly into swizzled B-fragment layout -------
    {
        int px = t & 15;
        int cs = t >> 4;                       // 0..15
        #pragma unroll
        for (int j = 0; j < 8; ++j) {
            int p  = cs + j * 16;              // channel pair 0..127
            int c0 = 2 * p;
            const float* gp = guide + ((b * C_IN + c0) * H_DIM + y) * W_DIM + x0 + px;
            unsigned int v = pack_bf2(gp[0], gp[H_DIM * W_DIM]);
            int kc = c0 >> 5, kk = c0 & 31;    // kk even
            int off = ((kc * 32) + swz_hk(kk) * 16 + px) * 16 + swz_i(kk);
            *(unsigned int*)&sGf[off] = v;
        }
    }
    __syncthreads();

    // ---- GEMM1: X[64x16] = (inv*W1)[64x256] * G[256x16] + bias1, ReLU ----
    if (wave < 4) {                 // wave-uniform branch: EXEC all ones
        int mt = wave;              // 16-row tile of M=64
        v8f acc = {};
        #pragma unroll
        for (int kc = 0; kc < C_IN / 32; ++kc) {
            Frag16 a, bb;
            const v4u* ap = (const v4u*)(w1f + ((mt * 8 + kc) * 32 + lane) * 16);
            a.q[0] = ap[0];  a.q[1] = ap[1];               // 2x global_load_b128
            const v4u* bp = (const v4u*)(sGf + (kc * 32 + lane) * 16);
            bb.q[0] = bp[0]; bb.q[1] = bp[1];              // 2x ds_load_b128
            acc = __builtin_amdgcn_wmma_f32_16x16x32_bf16(
                false, a.v, false, bb.v, (short)0, acc, false, false);
        }
        // bias + ReLU epilogue, packed pairwise into swizzled sXf
        #pragma unroll
        for (int r = 0; r < 8; r += 2) {
            int m0 = mt * 16 + hk * 8 + r;
            float v0 = fmaxf(acc[r]     + bias1[m0],     0.0f);
            float v1 = fmaxf(acc[r + 1] + bias1[m0 + 1], 0.0f);
            unsigned int pv = pack_bf2(v0, v1);
            int kc2 = m0 >> 5, kk = m0 & 31;   // kk even, pair in same octet
            int off = ((kc2 * 32) + swz_hk(kk) * 16 + ln) * 16 + swz_i(kk);
            *(unsigned int*)&sXf[off] = pv;
        }
    }
    __syncthreads();

    // ---- GEMM2: Wt[784x16] = W2[784x64] * X[64x16] + b2 ------------------
    for (int mt2 = wave; mt2 < K2G / 16; mt2 += 8) {   // 49 tiles over 8 waves
        v8f acc = {};
        #pragma unroll
        for (int kc = 0; kc < CR / 32; ++kc) {
            Frag16 a, bb;
            const v4u* ap = (const v4u*)(w2f + ((mt2 * 2 + kc) * 32 + lane) * 16);
            a.q[0] = ap[0];  a.q[1] = ap[1];
            const v4u* bp = (const v4u*)(sXf + (kc * 32 + lane) * 16);
            bb.q[0] = bp[0]; bb.q[1] = bp[1];
            acc = __builtin_amdgcn_wmma_f32_16x16x32_bf16(
                false, a.v, false, bb.v, (short)0, acc, false, false);
        }
        #pragma unroll
        for (int r = 0; r < 8; ++r) {
            int m = mt2 * 16 + hk * 8 + r;
            sW[m * TILE + ln] = acc[r] + b2[m];
        }
    }
    __syncthreads();

    // ---- dynamic 7x7 aggregation + residual ------------------------------
    {
        int px = t & 15;
        int g  = t >> 4;                 // group 0..15
        int xx = x0 + px;
        float accs[GC];
        #pragma unroll
        for (int j = 0; j < GC; ++j) accs[j] = 0.0f;

        #pragma unroll 1
        for (int k = 0; k < KK; ++k) {
            int dy = k / KSZ - 3;
            int dx = k % KSZ - 3;
            int yy = y + dy;
            int xc = xx + dx;
            float wgt = sW[(g * KK + k) * TILE + px];
            if (yy >= 0 && yy < H_DIM && xc >= 0 && xc < W_DIM) {
                const float* fp = fm + ((b * C_IN + g * GC) * H_DIM + yy) * W_DIM + xc;
                #pragma unroll
                for (int j = 0; j < GC; ++j)
                    accs[j] = fmaf(wgt, fp[j * H_DIM * W_DIM], accs[j]);
            }
        }

        const float* fc = fm  + ((b * C_IN + g * GC) * H_DIM + y) * W_DIM + xx;
        float*       op = out + ((b * C_IN + g * GC) * H_DIM + y) * W_DIM + xx;
        #pragma unroll
        for (int j = 0; j < GC; ++j)
            op[j * H_DIM * W_DIM] = accs[j] + fc[j * H_DIM * W_DIM];
    }
}

extern "C" void kernel_launch(void* const* d_in, const int* in_sizes, int n_in,
                              void* d_out, int out_size, void* d_ws, size_t ws_size,
                              hipStream_t stream) {
    const float* fm       = (const float*)d_in[0];
    const float* guide    = (const float*)d_in[1];
    const float* w1       = (const float*)d_in[2];
    const float* bn_gamma = (const float*)d_in[3];
    const float* bn_beta  = (const float*)d_in[4];
    const float* bn_mean  = (const float*)d_in[5];
    const float* bn_var   = (const float*)d_in[6];
    const float* w2       = (const float*)d_in[7];
    const float* b2       = (const float*)d_in[8];
    float* out = (float*)d_out;

    unsigned short* w1f = (unsigned short*)d_ws;           // 64*256  bf16 swizzled
    unsigned short* w2f = w1f + CR * C_IN;                 // 784*64  bf16 swizzled
    float* bias1 = (float*)(w2f + K2G * CR);               // [64] BN bias

    int prep_n = K2G * CR;                                 // 50176 (max of the two)
    prep_weights_swz<<<(prep_n + 255) / 256, 256, 0, stream>>>(
        w1, w2, bn_gamma, bn_beta, bn_mean, bn_var, w1f, w2f, bias1);

    int blocks = BATCH * H_DIM * (W_DIM / TILE);           // 1024
    cross_involution_fused<<<blocks, 256, 0, stream>>>(
        fm, guide, w1f, bias1, w2f, b2, out);
}
